// BahdanauAttention_54030688583862
// MI455X (gfx1250) — matrix operand order repaired and checked by reading, not compiled, not run on hardware
//
#include <hip/hip_runtime.h>
#include <hip/hip_bf16.h>
#include <math.h>

#define HDIM   512
#define LKDIM  4096
#define BDIM   32
#define TILE_L 128
#define LDA    520   // padded f16 row stride (+16B) to rotate LDS banks per row

typedef __attribute__((ext_vector_type(16))) _Float16 v16h;
typedef __attribute__((ext_vector_type(8)))  _Float16 f16x8;
typedef __attribute__((ext_vector_type(4)))  _Float16 f16x4;
typedef __attribute__((ext_vector_type(8)))  float    v8f;

// CDNA5 has a native TRANS-unit tanh (V_TANH_F32). Use the builtin when the
// toolchain declares it; otherwise emit the instruction directly.
__device__ __forceinline__ float fast_tanh(float x) {
#if __has_builtin(__builtin_amdgcn_tanhf)
    return __builtin_amdgcn_tanhf(x);
#elif __has_builtin(__builtin_amdgcn_tanh_f32)
    return __builtin_amdgcn_tanh_f32(x);
#else
    float y;
    asm volatile("v_tanh_f32 %0, %1" : "=v"(y) : "v"(x));
    return y;
#endif
}

// ---------------------------------------------------------------------------
// Kernel 1: q_proj[b,h] = query[b,:] . Wa_w[h,:] + Wa_b[h]   (tiny, fp32)
// ---------------------------------------------------------------------------
__global__ __launch_bounds__(64) void qproj_kernel(
    const float* __restrict__ query, const float* __restrict__ Wa_w,
    const float* __restrict__ Wa_b, float* __restrict__ qproj)
{
    const int b = blockIdx.y;
    const int h = blockIdx.x * 64 + threadIdx.x;
    const float4* q = (const float4*)(query + (size_t)b * HDIM);
    const float4* w = (const float4*)(Wa_w + (size_t)h * HDIM);
    float acc = Wa_b[h];
#pragma unroll 8
    for (int j = 0; j < HDIM / 4; ++j) {
        float4 qv = q[j], wv = w[j];
        acc += qv.x * wv.x + qv.y * wv.y + qv.z * wv.z + qv.w * wv.w;
    }
    qproj[(size_t)b * HDIM + h] = acc;
}

// ---------------------------------------------------------------------------
// Kernel 2: fused  scores[b,l] = Va_w . tanh(keys[b,l,:]@Ua_w.T + Ua_b + qproj[b,:])
// WMMA f16 GEMM with f32 accumulate; tanh+dot fused on C fragments.
// Block = 256 threads (8 waves), tile = 128 key rows of one batch.
// ---------------------------------------------------------------------------
__global__ __launch_bounds__(256) void scores_kernel(
    const float* __restrict__ keys, const float* __restrict__ Ua_w,
    const float* __restrict__ Ua_b, const float* __restrict__ Va_w,
    const float* __restrict__ qproj, float* __restrict__ scores)
{
    __shared__ _Float16 ldsA[TILE_L * LDA];  // 128x512 f16 keys tile (padded)
    __shared__ _Float16 ldsB[16 * LDA];      // 16x512  f16 Ua_w panel (padded)
    __shared__ float    qbv[HDIM];           // qproj[b,:] + Ua_b  (block-invariant)
    __shared__ float    vav[HDIM];           // Va_w

    const int b      = blockIdx.y;
    const int l0     = blockIdx.x * TILE_L;
    const int tid    = threadIdx.x;
    const int wave   = tid >> 5;
    const int lane   = tid & 31;
    const int half   = lane >> 4;   // 0: lanes 0-15, 1: lanes 16-31
    const int lane16 = lane & 15;

    // ---- stage block-invariant bias/weight vectors ----
    for (int i = tid; i < HDIM; i += 256) {
        qbv[i] = qproj[(size_t)b * HDIM + i] + Ua_b[i];
        vav[i] = Va_w[i];
    }

    // ---- stage A: keys[b, l0:l0+128, :] fp32 -> f16 LDS (coalesced float4) ----
    {
        const float4* src = (const float4*)(keys + ((size_t)b * LKDIM + l0) * HDIM);
        for (int i = tid; i < TILE_L * (HDIM / 4); i += 256) {
            const int row = i / (HDIM / 4);
            const int c4  = i % (HDIM / 4);
            float4 v = src[(size_t)row * (HDIM / 4) + c4];
            f16x4 hv = { (_Float16)v.x, (_Float16)v.y, (_Float16)v.z, (_Float16)v.w };
            *(f16x4*)&ldsA[row * LDA + c4 * 4] = hv;
        }
    }
    __syncthreads();

    float sAcc[8];
#pragma unroll
    for (int r = 0; r < 8; ++r) sAcc[r] = 0.0f;

    const int arow = wave * 16 + lane16;  // this wave's A-tile row for this lane

    for (int nt = 0; nt < HDIM / 16; ++nt) {
        const int h0 = nt * 16;
        // ---- stage B panel: Ua_w[h0:h0+16, :] fp32 -> f16 LDS, shared by 8 waves
        __syncthreads();
        {
            const float4* src = (const float4*)(Ua_w + (size_t)h0 * HDIM);
            for (int i = tid; i < 16 * (HDIM / 4); i += 256) {
                const int row = i / (HDIM / 4);
                const int c4  = i % (HDIM / 4);
                float4 v = src[(size_t)row * (HDIM / 4) + c4];
                f16x4 hv = { (_Float16)v.x, (_Float16)v.y, (_Float16)v.z, (_Float16)v.w };
                *(f16x4*)&ldsB[row * LDA + c4 * 4] = hv;
            }
        }
        __syncthreads();

        v8f acc = {};
#pragma unroll 4
        for (int kt = 0; kt < HDIM / 32; ++kt) {
            const int k0 = kt * 32;
            // A fragment 16x32 f16: lanes 0-15 hold K {0..7,16..23}, lanes 16-31 {8..15,24..31}
            union { v16h v; f16x8 h[2]; } af;
            const _Float16* ap = &ldsA[arow * LDA + k0 + (half << 3)];
            af.h[0] = *(const f16x8*)(ap);
            af.h[1] = *(const f16x8*)(ap + 16);
            // B fragment 32x16 f16: lane = output col n; lanes 0-15 K {0..7,16..23}, lanes 16-31 {8..15,24..31}
            union { v16h v; f16x8 h[2]; } bf;
            const _Float16* bp = &ldsB[lane16 * LDA + k0 + (half << 3)];
            bf.h[0] = *(const f16x8*)(bp);
            bf.h[1] = *(const f16x8*)(bp + 16);
            acc = __builtin_amdgcn_wmma_f32_16x16x32_f16(
                      /*neg_a=*/false, af.v, /*neg_b=*/false, bf.v,
                      /*c_mod=*/(short)0, acc, /*reuse_a=*/false, /*reuse_b=*/false);
        }

        // ---- fused epilogue on C fragment: +bias, tanh (TRANS unit), dot Va_w ----
        const int   h  = h0 + lane16;               // C layout: lane = column n
        const float qb = qbv[h];
        const float va = vav[h];
#pragma unroll
        for (int r = 0; r < 8; ++r)                 // VGPR r = row (half? r+8 : r)
            sAcc[r] += va * fast_tanh(acc[r] + qb);
    }

    // reduce over the 16 columns held by each half-wave
#pragma unroll
    for (int off = 1; off < 16; off <<= 1) {
#pragma unroll
        for (int r = 0; r < 8; ++r)
            sAcc[r] += __shfl_xor(sAcc[r], off, 32);
    }
    const int lbase = l0 + wave * 16 + half * 8;    // half 0 -> rows 0-7, half 1 -> rows 8-15
    if (lane16 == 0) {
#pragma unroll
        for (int r = 0; r < 8; ++r)
            scores[(size_t)b * LKDIM + lbase + r] = sAcc[r];
    }
}

// ---------------------------------------------------------------------------
// Kernel 3: per-batch masked softmax over Lk + context = weights . keys
// Block = 512 threads (16 waves), one block per batch.
// ---------------------------------------------------------------------------
__global__ __launch_bounds__(512) void softmax_context_kernel(
    const float* __restrict__ keys, const float* __restrict__ scores,
    const float* __restrict__ Va_b, const float* __restrict__ temp,
    const int* __restrict__ vlen,
    float* __restrict__ context, float* __restrict__ weights)
{
    __shared__ float wsm[LKDIM];
    __shared__ float red[16];
    const int b    = blockIdx.x;
    const int tid  = threadIdx.x;
    const int wave = tid >> 5, lane = tid & 31;
    const int nvalid = vlen[b];
    const float invT = 1.0f / temp[0];
    const float vb   = Va_b[0];

    // masked, temperature-scaled scores + running max
    float m = -3.4e38f;
    for (int l = tid; l < LKDIM; l += 512) {
        float s = (l < nvalid) ? (scores[(size_t)b * LKDIM + l] + vb) * invT : -1e9f;
        wsm[l] = s;
        m = fmaxf(m, s);
    }
#pragma unroll
    for (int off = 16; off > 0; off >>= 1) m = fmaxf(m, __shfl_xor(m, off, 32));
    if (lane == 0) red[wave] = m;
    __syncthreads();
    if (wave == 0) {
        float t = (lane < 16) ? red[lane] : -3.4e38f;
#pragma unroll
        for (int off = 16; off > 0; off >>= 1) t = fmaxf(t, __shfl_xor(t, off, 32));
        if (lane == 0) red[0] = t;
    }
    __syncthreads();
    m = red[0];
    __syncthreads();

    float sum = 0.0f;
    for (int l = tid; l < LKDIM; l += 512) {
        float e = __expf(wsm[l] - m);
        wsm[l] = e;
        sum += e;
    }
#pragma unroll
    for (int off = 16; off > 0; off >>= 1) sum += __shfl_xor(sum, off, 32);
    if (lane == 0) red[wave] = sum;
    __syncthreads();
    if (wave == 0) {
        float t = (lane < 16) ? red[lane] : 0.0f;
#pragma unroll
        for (int off = 16; off > 0; off >>= 1) t += __shfl_xor(t, off, 32);
        if (lane == 0) red[0] = t;
    }
    __syncthreads();
    const float inv = 1.0f / red[0];

    for (int l = tid; l < LKDIM; l += 512) {
        float w = wsm[l] * inv;
        wsm[l] = w;
        weights[(size_t)b * LKDIM + l] = w;
    }
    __syncthreads();

    // context[b,h] = sum_l w[l] * keys[b,l,h]; thread tid -> h (coalesced rows)
    const float* kb = keys + (size_t)b * LKDIM * HDIM + tid;
    float acc = 0.0f;
#pragma unroll 8
    for (int l = 0; l < LKDIM; ++l)
        acc += wsm[l] * kb[(size_t)l * HDIM];
    context[(size_t)b * HDIM + tid] = acc;
}

// ---------------------------------------------------------------------------
extern "C" void kernel_launch(void* const* d_in, const int* in_sizes, int n_in,
                              void* d_out, int out_size, void* d_ws, size_t ws_size,
                              hipStream_t stream)
{
    const float* query = (const float*)d_in[0];
    const float* keys  = (const float*)d_in[1];
    const float* Wa_w  = (const float*)d_in[2];
    const float* Wa_b  = (const float*)d_in[3];
    const float* Ua_w  = (const float*)d_in[4];
    const float* Ua_b  = (const float*)d_in[5];
    const float* Va_w  = (const float*)d_in[6];
    const float* Va_b  = (const float*)d_in[7];
    const float* temp  = (const float*)d_in[8];
    const int*   vlen  = (const int*)d_in[9];

    float* ctx_out = (float*)d_out;              // [B,1,H]  (first output)
    float* w_out   = ctx_out + BDIM * HDIM;      // [B,1,Lk] (second output)

    float* qproj  = (float*)d_ws;                // B*H   fp32
    float* scores = qproj + BDIM * HDIM;         // B*Lk  fp32

    qproj_kernel<<<dim3(HDIM / 64, BDIM), 64, 0, stream>>>(query, Wa_w, Wa_b, qproj);
    scores_kernel<<<dim3(LKDIM / TILE_L, BDIM), 256, 0, stream>>>(
        keys, Ua_w, Ua_b, Va_w, qproj, scores);
    softmax_context_kernel<<<BDIM, 512, 0, stream>>>(
        keys, scores, Va_b, temp, vlen, ctx_out, w_out);
}
